// RNNDecoder_14362370638285
// MI455X (gfx1250) — compile-verified, matrix operand order checked by reading, not run on hardware
//
#include <hip/hip_runtime.h>
#include <hip/hip_bf16.h>

typedef __bf16 bf16;
typedef bf16 v16bf __attribute__((ext_vector_type(16)));
typedef bf16 v8bf  __attribute__((ext_vector_type(8)));
typedef bf16 v4bf  __attribute__((ext_vector_type(4)));
typedef float v8f  __attribute__((ext_vector_type(8)));

#define B_SZ   256
#define H_SZ   1024
#define T_LEN  128
#define MID_SZ 512
#define KC     32

__device__ __forceinline__ float sigf(float x) { return 1.0f / (1.0f + __expf(-x)); }

// ---------------- preamble kernels ----------------
__global__ void cvt_bf16_kernel(bf16* __restrict__ dst, const float* __restrict__ src, int n) {
  int i = blockIdx.x * blockDim.x + threadIdx.x;
  if (i < n) dst[i] = (bf16)src[i];
}

// Build concatenated layer-1 weights: Wcat[n][k] with k<1024 -> w_ih1, else w_hh1 (bf16)
__global__ void cat_w1_kernel(bf16* __restrict__ dst, const float* __restrict__ wih,
                              const float* __restrict__ whh) {
  int i = blockIdx.x * blockDim.x + threadIdx.x;  // 4096*2048 total
  int nn = i >> 11;
  int k  = i & 2047;
  float v = (k < 1024) ? wih[nn * 1024 + k] : whh[nn * 1024 + (k - 1024)];
  dst[i] = (bf16)v;
}

__global__ void bias_init_kernel(float* __restrict__ b0, const float* __restrict__ bi0,
                                 const float* __restrict__ bh0,
                                 float* __restrict__ b1, const float* __restrict__ bi1,
                                 const float* __restrict__ bh1, float* __restrict__ inp) {
  int i = blockIdx.x * blockDim.x + threadIdx.x;
  if (i < 4 * H_SZ) { b0[i] = bi0[i] + bh0[i]; b1[i] = bi1[i] + bh1[i]; }
  if (i < B_SZ) inp[i] = 0.0f;
}

// ---------------- fused LSTM layer (GEMM + cell nonlinearity) ----------------
// Block: 256 threads = 8 waves (2 M-waves x 32 rows, 4 N-waves x 16 cols per gate).
// Tile: BM=64 rows, BN=64 cols per gate; grid (H/64, B/64).
// Each B fragment is reused for two A fragments: 8 WMMA per chunk from 6 LDS frag loads.
// Global->LDS staging is software-pipelined through registers.
__global__ __launch_bounds__(256)
void lstm_layer_kernel(const bf16* __restrict__ W, const float* __restrict__ hA0,
                       const float* __restrict__ hA1, const float* __restrict__ bias,
                       const float* __restrict__ wih0, const float* __restrict__ inp,
                       float* __restrict__ cState, float* __restrict__ hOut,
                       int Ktot, int useInp) {
  __shared__ __attribute__((aligned(64))) bf16 As[64][KC];      // 4 KB
  __shared__ __attribute__((aligned(64))) bf16 Bs[4][64][KC];   // 16 KB

  const int tid   = threadIdx.x;
  const int lane  = tid & 31, lrow = lane & 15, lhi = lane >> 4;
  const int wid   = tid >> 5;
  const int waveM = wid & 1;    // 2 M groups of 32 rows
  const int waveN = wid >> 1;   // 4 N groups of 16 cols (per gate)
  const int mBase = blockIdx.y * 64;
  const int nBase = blockIdx.x * 64;

  v8f acc[4][2];  // [gate][m-frag]
#pragma unroll
  for (int g = 0; g < 4; ++g)
#pragma unroll
    for (int mf = 0; mf < 2; ++mf) acc[g][mf] = {};

  // loader mapping (independent of compute mapping)
  const int bg   = tid >> 6;       // gate slab 0..3
  const int bn   = tid & 63;       // column 0..63
  const int arow = tid >> 2;       // 0..63
  const int akk  = (tid & 3) * 8;

  const bf16* wRow = W + (size_t)(bg * H_SZ + nBase + bn) * Ktot;

  uint4  bReg[4];
  float4 aReg[2];
  auto stage = [&](int k0) {
    const uint4* wv = (const uint4*)(wRow + k0);
    bReg[0] = wv[0]; bReg[1] = wv[1]; bReg[2] = wv[2]; bReg[3] = wv[3];
    __builtin_prefetch(wRow + k0 + 2 * KC, 0, 0);  // global_prefetch for chunk after next
    const int kg = k0 + akk;
    const float* src = (kg < H_SZ) ? hA0 : hA1;    // layer1: [h0_new, h1_prev]
    const float* ap  = src + (size_t)(mBase + arow) * H_SZ + (kg & (H_SZ - 1));
    aReg[0] = ((const float4*)ap)[0];
    aReg[1] = ((const float4*)ap)[1];
  };

  stage(0);
  for (int k0 = 0; k0 < Ktot; k0 += KC) {
    {  // commit staged registers to LDS
      uint4* bd = (uint4*)(&Bs[bg][bn][0]);
      bd[0] = bReg[0]; bd[1] = bReg[1]; bd[2] = bReg[2]; bd[3] = bReg[3];
      v8bf t;
      t[0] = (bf16)aReg[0].x; t[1] = (bf16)aReg[0].y; t[2] = (bf16)aReg[0].z; t[3] = (bf16)aReg[0].w;
      t[4] = (bf16)aReg[1].x; t[5] = (bf16)aReg[1].y; t[6] = (bf16)aReg[1].z; t[7] = (bf16)aReg[1].w;
      *(v8bf*)(&As[arow][akk]) = t;
    }
    __syncthreads();
    if (k0 + KC < Ktot) stage(k0 + KC);  // prefetch next chunk while computing this one

    v16bf a0 = *(const v16bf*)(&As[waveM * 32 + lrow][lhi * 16]);
    v16bf a1 = *(const v16bf*)(&As[waveM * 32 + 16 + lrow][lhi * 16]);
#pragma unroll
    for (int g = 0; g < 4; ++g) {
      v16bf b = *(const v16bf*)(&Bs[g][waveN * 16 + lrow][lhi * 16]);
      acc[g][0] = __builtin_amdgcn_wmma_f32_16x16x32_bf16(
          false, a0, false, b, (short)0, acc[g][0], false, false);
      acc[g][1] = __builtin_amdgcn_wmma_f32_16x16x32_bf16(
          false, a1, false, b, (short)0, acc[g][1], false, false);
    }
    __syncthreads();
  }

  // ---- fused LSTM cell epilogue: C-fragment layout row = r + 8*lhi, col = lane%16 ----
  const int n = nBase + waveN * 16 + lrow;
  const float bi  = bias[n];
  const float bff = bias[H_SZ + n];
  const float bgg = bias[2 * H_SZ + n];
  const float boo = bias[3 * H_SZ + n];
  float wi = 0.f, wf = 0.f, wg = 0.f, wo = 0.f;
  if (useInp) {
    wi = wih0[n]; wf = wih0[H_SZ + n]; wg = wih0[2 * H_SZ + n]; wo = wih0[3 * H_SZ + n];
  }
#pragma unroll
  for (int mf = 0; mf < 2; ++mf) {
#pragma unroll
    for (int r = 0; r < 8; ++r) {
      const int b = mBase + waveM * 32 + mf * 16 + r + lhi * 8;
      const float x = useInp ? inp[b] : 0.0f;
      const float gi = acc[0][mf][r] + bi  + x * wi;
      const float gf = acc[1][mf][r] + bff + x * wf;
      const float gg = acc[2][mf][r] + bgg + x * wg;
      const float go = acc[3][mf][r] + boo + x * wo;
      const size_t idx = (size_t)b * H_SZ + n;
      const float cp = cState[idx];
      const float cn = sigf(gf) * cp + sigf(gi) * tanhf(gg);
      const float hn = sigf(go) * tanhf(cn);
      cState[idx] = cn;
      hOut[idx]   = hn;
    }
  }
}

// ---------------- head: mid = relu(relu(h1) @ W1^T + b1); out = mid @ w2^T + b2 ----------------
// Block: 256 threads = 8 waves, each covering the full 32 rows (2 M-frags) x 64 cols.
// BM=32, N=512 full per block; grid = B/32 = 8.
__global__ __launch_bounds__(256)
void head_kernel(const bf16* __restrict__ LW1, const float* __restrict__ h1,
                 const float* __restrict__ b1, const float* __restrict__ w2,
                 const float* __restrict__ b2, float* __restrict__ out,
                 float* __restrict__ inpNext, int t) {
  __shared__ __attribute__((aligned(64))) bf16 As[32][KC];    // 2 KB
  __shared__ __attribute__((aligned(64))) bf16 Bs[512][KC];   // 32 KB
  __shared__ float rowsum[32];

  const int tid   = threadIdx.x;
  const int lane  = tid & 31, lrow = lane & 15, lhi = lane >> 4;
  const int waveN = tid >> 5;       // 8 N groups of 64 cols
  const int mBase = blockIdx.x * 32;

  if (tid < 32) rowsum[tid] = 0.0f;

  v8f acc[4][2];  // [n-frag][m-frag]
#pragma unroll
  for (int nf = 0; nf < 4; ++nf)
#pragma unroll
    for (int mf = 0; mf < 2; ++mf) acc[nf][mf] = {};

  const int arow = tid >> 3;       // 0..31
  const int akk  = (tid & 7) * 4;

  uint4  bReg[8];
  float4 aReg;
  auto stage = [&](int k0) {
#pragma unroll
    for (int rep = 0; rep < 2; ++rep) {
      const int nn = tid + rep * 256;
      const uint4* wv = (const uint4*)(LW1 + (size_t)nn * H_SZ + k0);
      bReg[rep * 4 + 0] = wv[0]; bReg[rep * 4 + 1] = wv[1];
      bReg[rep * 4 + 2] = wv[2]; bReg[rep * 4 + 3] = wv[3];
    }
    aReg = *(const float4*)(h1 + (size_t)(mBase + arow) * H_SZ + k0 + akk);
  };

  stage(0);
  for (int k0 = 0; k0 < H_SZ; k0 += KC) {
    {  // commit staged registers to LDS (relu fused into A conversion)
#pragma unroll
      for (int rep = 0; rep < 2; ++rep) {
        uint4* bd = (uint4*)(&Bs[tid + rep * 256][0]);
        bd[0] = bReg[rep * 4 + 0]; bd[1] = bReg[rep * 4 + 1];
        bd[2] = bReg[rep * 4 + 2]; bd[3] = bReg[rep * 4 + 3];
      }
      v4bf tv;
      tv[0] = (bf16)fmaxf(aReg.x, 0.f); tv[1] = (bf16)fmaxf(aReg.y, 0.f);
      tv[2] = (bf16)fmaxf(aReg.z, 0.f); tv[3] = (bf16)fmaxf(aReg.w, 0.f);
      *(v4bf*)(&As[arow][akk]) = tv;
    }
    __syncthreads();
    if (k0 + KC < H_SZ) stage(k0 + KC);

    v16bf a0 = *(const v16bf*)(&As[lrow][lhi * 16]);
    v16bf a1 = *(const v16bf*)(&As[16 + lrow][lhi * 16]);
#pragma unroll
    for (int nf = 0; nf < 4; ++nf) {
      v16bf b = *(const v16bf*)(&Bs[waveN * 64 + nf * 16 + lrow][lhi * 16]);
      acc[nf][0] = __builtin_amdgcn_wmma_f32_16x16x32_bf16(
          false, a0, false, b, (short)0, acc[nf][0], false, false);
      acc[nf][1] = __builtin_amdgcn_wmma_f32_16x16x32_bf16(
          false, a1, false, b, (short)0, acc[nf][1], false, false);
    }
    __syncthreads();
  }

  // relu(+b1) * w2, reduce over N: lanes (shfl) -> waves (LDS atomic)
#pragma unroll
  for (int nf = 0; nf < 4; ++nf) {
    const int n = waveN * 64 + nf * 16 + lrow;
    const float bb = b1[n], ww = w2[n];
#pragma unroll
    for (int mf = 0; mf < 2; ++mf) {
#pragma unroll
      for (int r = 0; r < 8; ++r) {
        float v = fmaxf(acc[nf][mf][r] + bb, 0.0f) * ww;
        v += __shfl_xor(v, 1, 32);
        v += __shfl_xor(v, 2, 32);
        v += __shfl_xor(v, 4, 32);
        v += __shfl_xor(v, 8, 32);
        if (lrow == 0) atomicAdd(&rowsum[mf * 16 + r + lhi * 8], v);
      }
    }
  }
  __syncthreads();
  if (tid < 32) {
    const int b = mBase + tid;
    const float o = rowsum[tid] + b2[0];
    out[(size_t)b * T_LEN + t] = o;   // (B, T, 1)
    inpNext[b] = o;                   // autoregressive feedback
  }
}

// ---------------- host launcher ----------------
extern "C" void kernel_launch(void* const* d_in, const int* in_sizes, int n_in,
                              void* d_out, int out_size, void* d_ws, size_t ws_size,
                              hipStream_t stream) {
  (void)in_sizes; (void)n_in; (void)out_size; (void)ws_size;
  const float* h_n    = (const float*)d_in[0];
  const float* c_n    = (const float*)d_in[1];
  const float* w_ih0  = (const float*)d_in[2];
  const float* w_hh0  = (const float*)d_in[3];
  const float* b_ih0  = (const float*)d_in[4];
  const float* b_hh0  = (const float*)d_in[5];
  const float* w_ih1  = (const float*)d_in[6];
  const float* w_hh1  = (const float*)d_in[7];
  const float* b_ih1  = (const float*)d_in[8];
  const float* b_hh1  = (const float*)d_in[9];
  const float* lin_w1 = (const float*)d_in[10];
  const float* lin_b1 = (const float*)d_in[11];
  const float* lin_w2 = (const float*)d_in[12];
  const float* lin_b2 = (const float*)d_in[13];

  char* p = (char*)d_ws;
  auto alloc = [&](size_t bytes) -> void* {
    void* r = (void*)p;
    p += (bytes + 255) & ~(size_t)255;
    return r;
  };
  bf16* W0   = (bf16*)alloc((size_t)4 * H_SZ * H_SZ * sizeof(bf16));       // 8 MB
  bf16* W1   = (bf16*)alloc((size_t)4 * H_SZ * 2 * H_SZ * sizeof(bf16));   // 16 MB
  bf16* LW1b = (bf16*)alloc((size_t)MID_SZ * H_SZ * sizeof(bf16));         // 1 MB
  float* bias0 = (float*)alloc(4 * H_SZ * sizeof(float));
  float* bias1 = (float*)alloc(4 * H_SZ * sizeof(float));
  const size_t stN = (size_t)B_SZ * H_SZ;
  float* h0buf[2] = { (float*)alloc(stN * 4), (float*)alloc(stN * 4) };
  float* h1buf[2] = { (float*)alloc(stN * 4), (float*)alloc(stN * 4) };
  float* c0  = (float*)alloc(stN * 4);
  float* c1  = (float*)alloc(stN * 4);
  float* inp = (float*)alloc(B_SZ * 4);

  // --- preamble (re-done every call for determinism) ---
  cvt_bf16_kernel<<<(4 * H_SZ * H_SZ) / 256, 256, 0, stream>>>(W0, w_hh0, 4 * H_SZ * H_SZ);
  cat_w1_kernel<<<(4 * H_SZ * 2 * H_SZ) / 256, 256, 0, stream>>>(W1, w_ih1, w_hh1);
  cvt_bf16_kernel<<<(MID_SZ * H_SZ) / 256, 256, 0, stream>>>(LW1b, lin_w1, MID_SZ * H_SZ);
  bias_init_kernel<<<16, 256, 0, stream>>>(bias0, b_ih0, b_hh0, bias1, b_ih1, b_hh1, inp);
  hipMemcpyAsync(h0buf[0], h_n,       stN * 4, hipMemcpyDeviceToDevice, stream);
  hipMemcpyAsync(h1buf[0], h_n + stN, stN * 4, hipMemcpyDeviceToDevice, stream);
  hipMemcpyAsync(c0,       c_n,       stN * 4, hipMemcpyDeviceToDevice, stream);
  hipMemcpyAsync(c1,       c_n + stN, stN * 4, hipMemcpyDeviceToDevice, stream);

  float* outF = (float*)d_out;
  dim3 lgrid(H_SZ / 64, B_SZ / 64);  // (16, 4)
  for (int t = 0; t < T_LEN; ++t) {
    const int r = t & 1, w = r ^ 1;
    // layer 0: K=1024, rank-1 input term in epilogue
    lstm_layer_kernel<<<lgrid, 256, 0, stream>>>(W0, h0buf[r], h0buf[r], bias0,
                                                 w_ih0, inp, c0, h0buf[w], H_SZ, 1);
    // layer 1: K=2048 over [h0_new, h1_prev]
    lstm_layer_kernel<<<lgrid, 256, 0, stream>>>(W1, h0buf[w], h1buf[r], bias1,
                                                 nullptr, nullptr, c1, h1buf[w], 2 * H_SZ, 0);
    // head + autoregressive feedback
    head_kernel<<<B_SZ / 32, 256, 0, stream>>>(LW1b, h1buf[w], lin_b1, lin_w2, lin_b2,
                                               outF, inp, t);
  }
}
